// NeuralNetwork_15444702396521
// MI455X (gfx1250) — compile-verified, matrix operand order checked by reading, not run on hardware
//
#include <hip/hip_runtime.h>
#include <hip/hip_bf16.h>
#include <stdint.h>

#define N_RAYS   32768
#define N_POINTS 200

typedef _Float16 v16h __attribute__((ext_vector_type(16)));
typedef float    v8f  __attribute__((ext_vector_type(8)));

union H16 { _Float16 f; unsigned short u; };
__device__ __forceinline__ unsigned short f2h(float x){ H16 h; h.f = (_Float16)x; return h.u; }

// Level table offsets (rows): level params are res = 16<<l,
// dense for l<3 (n = 4920, 35944, 274632), hashed n = 2^19 for l>=3.
__device__ __forceinline__ uint32_t level_offset(int l){
  return (l==0)?0u:(l==1)?4920u:(l==2)?40864u:(l==3)?315496u:(l==4)?839784u:1364072u;
}

// ---------------------------------------------------------------------------
// Kernel 0: pad + convert MLP weights to f16 (K padded to 32 for layer1,
// out-rows padded to 16 for layer4) so WMMA tiles need no masking.
// ---------------------------------------------------------------------------
__global__ void __launch_bounds__(256) k_prep(
    const float* __restrict__ w1, const float* __restrict__ w2,
    const float* __restrict__ w3, const float* __restrict__ w4,
    _Float16* __restrict__ W1p, _Float16* __restrict__ W2p,
    _Float16* __restrict__ W3p, _Float16* __restrict__ W4p)
{
  int t = blockIdx.x * blockDim.x + threadIdx.x;
  if (t < 2048) {                       // W1p: 64 x 32 (w1 is [64,27])
    int r = t >> 5, k = t & 31;
    W1p[t] = (_Float16)((k < 27) ? w1[r*27 + k] : 0.0f);
  } else if (t < 2048 + 4096) {         // W2p: 64 x 64
    int i = t - 2048;  W2p[i] = (_Float16)w2[i];
  } else if (t < 2048 + 8192) {         // W3p: 64 x 64
    int i = t - 6144;  W3p[i] = (_Float16)w3[i];
  } else if (t < 2048 + 8192 + 1024) {  // W4p: 16 x 64 (w4 is [3,64])
    int i = t - 10240;
    W4p[i] = (_Float16)((i < 3*64) ? w4[i] : 0.0f);
  }
}

// ---------------------------------------------------------------------------
// Kernel 1: one wave (32 lanes) per ray. Each lane handles points
// p = lane, lane+32, ... (200 points). Per point: 6-level hash-grid encode
// (48 gathers from the 7.5MB label table -> fully L2 resident), sigmoid,
// then a wave32 shfl_xor reduction for max and first-hit argmax.
// Lane 0 emits output_hits, indices_first, and the hit point + ray dir.
// ---------------------------------------------------------------------------
__global__ void __launch_bounds__(256) k_label(
    const float* __restrict__ x, const float* __restrict__ tblL,
    const float* __restrict__ wl, const float* __restrict__ blb,
    float* __restrict__ outHits, float* __restrict__ outSig,
    int* __restrict__ outIdx, float* __restrict__ ptsdirs)
{
  const int lane = threadIdx.x & 31;
  const int ray  = (blockIdx.x * blockDim.x + threadIdx.x) >> 5;

  const float4 ang = reinterpret_cast<const float4*>(x)[ray];
  float st1 = __sinf(ang.x), ct1 = __cosf(ang.x);
  float st2 = __sinf(ang.z), ct2 = __cosf(ang.z);
  float p1x = st1*__cosf(ang.y), p1y = st1*__sinf(ang.y), p1z = ct1;
  float p2x = st2*__cosf(ang.w), p2y = st2*__sinf(ang.w), p2z = ct2;
  float dx = p2x - p1x, dy = p2y - p1y, dz = p2z - p1z;
  float len = sqrtf(dx*dx + dy*dy + dz*dz);
  len = fminf(fmaxf(len, 1e-6f), 1e6f);
  float invLen = 1.0f / len;

  float wv[6];
  #pragma unroll
  for (int l = 0; l < 6; ++l) wv[l] = wl[l];
  const float bias = blb[0];

  float vmax  = -1e30f;
  int   first = 0x7fffffff;

  for (int p = lane; p < N_POINTS; p += 32) {
    float t = (float)p * (1.0f / 199.0f);
    float x01x = (p1x + dx*t + 1.0f) * 0.5f;
    float x01y = (p1y + dy*t + 1.0f) * 0.5f;
    float x01z = (p1z + dz*t + 1.0f) * 0.5f;

    float logit = bias;
    #pragma unroll
    for (int l = 0; l < 6; ++l) {
      const uint32_t res   = 16u << l;
      const float    scale = (float)(res - 1u);
      float psx = x01x*scale + 0.5f; float gx = floorf(psx); float fx = psx - gx;
      float psy = x01y*scale + 0.5f; float gy = floorf(psy); float fy = psy - gy;
      float psz = x01z*scale + 0.5f; float gz = floorf(psz); float fz = psz - gz;
      uint32_t ix = (uint32_t)(int)gx, iy = (uint32_t)(int)gy, iz = (uint32_t)(int)gz;
      float wx0 = 1.0f - fx, wy0 = 1.0f - fy, wz0 = 1.0f - fz;
      const float* tb = tblL + level_offset(l);

      float feat = 0.0f;
      #pragma unroll
      for (int c = 0; c < 8; ++c) {
        uint32_t bx = c & 1u, by = (c >> 1) & 1u, bz = (c >> 2) & 1u;
        float w = (bx ? fx : wx0) * (by ? fy : wy0) * (bz ? fz : wz0);
        uint32_t cx = ix + bx, cy = iy + by, cz = iz + bz;
        uint32_t idx;
        if (l < 3) {              // dense: idx < n, so the '% n' is identity
          const uint32_t S = res + 1u;
          idx = cx + cy*S + cz*S*S;
        } else {                  // hashed: n = 2^19 -> mod == AND
          idx = cx ^ (cy * 2654435761u) ^ (cz * 805459861u);
          idx &= 524287u;
        }
        feat = fmaf(w, tb[idx], feat);
      }
      logit = fmaf(feat, wv[l], logit);
    }

    float sg = 1.0f / (1.0f + __expf(-logit));
    outSig[ray * N_POINTS + p] = sg;            // coalesced (p = lane + 32*i)
    vmax = fmaxf(vmax, sg);
    if (sg > 0.5f && p < first) first = p;
  }

  // wave32 reduction: max(sigmoid) and min(first-hit index)
  #pragma unroll
  for (int m = 16; m >= 1; m >>= 1) {
    float ov = __shfl_xor(vmax, m, 32);
    vmax = fmaxf(vmax, ov);
    int oi = __shfl_xor(first, m, 32);
    first = (oi < first) ? oi : first;
  }

  if (lane == 0) {
    int idx = (first == 0x7fffffff) ? 0 : first; // argmax of all-zero hits == 0
    outHits[ray] = vmax;
    outIdx[ray]  = idx;
    float t = (float)idx * (1.0f / 199.0f);
    float* pd = ptsdirs + ray * 8;
    pd[0] = p1x + dx*t; pd[1] = p1y + dy*t; pd[2] = p1z + dz*t;
    pd[3] = dx * invLen; pd[4] = dy * invLen; pd[5] = dz * invLen;
  }
}

// ---------------------------------------------------------------------------
// Kernel 2: one thread per ray. Encode the hit point with the 4-channel RGB
// table (float4 gathers, table is 30MB -> L2 resident), assemble the padded
// f16 MLP input row [dir(3) | feat(24) | 0(5)] in WMMA-friendly row-major.
// ---------------------------------------------------------------------------
__global__ void __launch_bounds__(256) k_rgb(
    const float* __restrict__ tblR, const float* __restrict__ ptsdirs,
    unsigned int* __restrict__ x16u)
{
  int r = blockIdx.x * blockDim.x + threadIdx.x;
  const float* pd = ptsdirs + r * 8;
  float h[32];
  h[0] = pd[3]; h[1] = pd[4]; h[2] = pd[5];
  float x01x = (pd[0] + 1.0f) * 0.5f;
  float x01y = (pd[1] + 1.0f) * 0.5f;
  float x01z = (pd[2] + 1.0f) * 0.5f;
  const float4* t4 = reinterpret_cast<const float4*>(tblR);

  #pragma unroll
  for (int l = 0; l < 6; ++l) {
    const uint32_t res   = 16u << l;
    const float    scale = (float)(res - 1u);
    float psx = x01x*scale + 0.5f; float gx = floorf(psx); float fx = psx - gx;
    float psy = x01y*scale + 0.5f; float gy = floorf(psy); float fy = psy - gy;
    float psz = x01z*scale + 0.5f; float gz = floorf(psz); float fz = psz - gz;
    uint32_t ix = (uint32_t)(int)gx, iy = (uint32_t)(int)gy, iz = (uint32_t)(int)gz;
    float wx0 = 1.0f - fx, wy0 = 1.0f - fy, wz0 = 1.0f - fz;
    const uint32_t off = level_offset(l);

    float ax = 0.f, ay = 0.f, az = 0.f, aw = 0.f;
    #pragma unroll
    for (int c = 0; c < 8; ++c) {
      uint32_t bx = c & 1u, by = (c >> 1) & 1u, bz = (c >> 2) & 1u;
      float w = (bx ? fx : wx0) * (by ? fy : wy0) * (bz ? fz : wz0);
      uint32_t cx = ix + bx, cy = iy + by, cz = iz + bz;
      uint32_t idx;
      if (l < 3) { const uint32_t S = res + 1u; idx = cx + cy*S + cz*S*S; }
      else       { idx = (cx ^ (cy*2654435761u) ^ (cz*805459861u)) & 524287u; }
      float4 v = t4[off + idx];
      ax = fmaf(w, v.x, ax); ay = fmaf(w, v.y, ay);
      az = fmaf(w, v.z, az); aw = fmaf(w, v.w, aw);
    }
    h[3 + l*4 + 0] = ax; h[3 + l*4 + 1] = ay;
    h[3 + l*4 + 2] = az; h[3 + l*4 + 3] = aw;
  }
  #pragma unroll
  for (int k = 27; k < 32; ++k) h[k] = 0.0f;

  unsigned int* dst = x16u + r * 16;     // 32 halves packed as 16 dwords
  #pragma unroll
  for (int i = 0; i < 16; ++i)
    dst[i] = (unsigned)f2h(h[2*i]) | ((unsigned)f2h(h[2*i + 1]) << 16);
}

// ---------------------------------------------------------------------------
// Kernel 3: WMMA MLP. One wave per 16-ray tile, 4 column-tiles of the 64
// hidden units. Between layers the f32 C-layout (row striped across lanes)
// is bounced through LDS as f16 to rebuild the A-layout (K striped across
// lanes), synchronized with s_wait_dscnt.
// A as loaded here (16x32 f16, contiguous K): lane L holds row (L&15), K = (L>>4)*16 .. +15.
// B as loaded here (32x16 f16, contiguous K): lane L holds col (L&15), K = (L>>4)*16 .. +15.
// C layout (16x16 f32): VGPR i, lane L -> row i + (L>>4)*8, col (L&15).
// ---------------------------------------------------------------------------
__global__ void __launch_bounds__(256) k_mlp(
    const _Float16* __restrict__ x16,
    const _Float16* __restrict__ W1p, const _Float16* __restrict__ W2p,
    const _Float16* __restrict__ W3p, const _Float16* __restrict__ W4p,
    const float* __restrict__ b1, const float* __restrict__ b2,
    const float* __restrict__ b3, const float* __restrict__ b4,
    float* __restrict__ rgbOut)
{
  __shared__ __align__(64) _Float16 lds[8][16 * 64];
  const int lane      = threadIdx.x & 31;
  const int waveInBlk = threadIdx.x >> 5;
  const int wave      = blockIdx.x * 8 + waveInBlk;
  const int rayBase   = wave * 16;
  const int mrow  = lane & 15;
  const int khalf = (lane >> 4) * 16;     // 0 or 16 (K sub-block within lane)
  const int rbase = (lane >> 4) * 8;      // C-layout row base for this lane
  _Float16* h16 = lds[waveInBlk];

  // ---- Layer 1: [16x32] x [32x64] ----
  v16h a = *reinterpret_cast<const v16h*>(x16 + (rayBase + mrow) * 32 + khalf);
  v8f acc[4];
  #pragma unroll
  for (int ct = 0; ct < 4; ++ct) {
    int c = ct * 16 + mrow;
    v16h bm = *reinterpret_cast<const v16h*>(W1p + c * 32 + khalf);
    v8f z = {};
    acc[ct] = __builtin_amdgcn_wmma_f32_16x16x32_f16(false, a, false, bm,
                                                     (short)0, z, false, false);
  }
  #pragma unroll
  for (int ct = 0; ct < 4; ++ct) {
    int c = ct * 16 + mrow;
    float bb = b1[c];
    #pragma unroll
    for (int i = 0; i < 8; ++i) {
      float v = acc[ct][i] + bb;
      v = v > 0.0f ? v : 0.0f;
      h16[(rbase + i) * 64 + c] = (_Float16)v;   // ds_store_b16
    }
  }
  asm volatile("s_wait_dscnt 0" ::: "memory");

  // ---- Layers 2 & 3: [16x64] x [64x64], K = 2 x 32 ----
  #pragma unroll
  for (int layer = 0; layer < 2; ++layer) {
    const _Float16* Wp = (layer == 0) ? W2p : W3p;
    const float*    Bp = (layer == 0) ? b2  : b3;
    v16h a0 = *reinterpret_cast<const v16h*>(h16 + mrow * 64 + khalf);
    v16h a1 = *reinterpret_cast<const v16h*>(h16 + mrow * 64 + 32 + khalf);
    v8f acc2[4];
    #pragma unroll
    for (int ct = 0; ct < 4; ++ct) {
      int c = ct * 16 + mrow;
      v16h bm0 = *reinterpret_cast<const v16h*>(Wp + c * 64 + khalf);
      v16h bm1 = *reinterpret_cast<const v16h*>(Wp + c * 64 + 32 + khalf);
      v8f z = {};
      v8f t0 = __builtin_amdgcn_wmma_f32_16x16x32_f16(false, a0, false, bm0,
                                                      (short)0, z, false, false);
      acc2[ct] = __builtin_amdgcn_wmma_f32_16x16x32_f16(false, a1, false, bm1,
                                                        (short)0, t0, false, false);
    }
    #pragma unroll
    for (int ct = 0; ct < 4; ++ct) {
      int c = ct * 16 + mrow;
      float bb = Bp[c];
      #pragma unroll
      for (int i = 0; i < 8; ++i) {
        float v = acc2[ct][i] + bb;
        v = v > 0.0f ? v : 0.0f;
        h16[(rbase + i) * 64 + c] = (_Float16)v;
      }
    }
    asm volatile("s_wait_dscnt 0" ::: "memory");
  }

  // ---- Layer 4: [16x64] x [64x16] (cols 3..15 zero-padded) ----
  {
    v16h a0 = *reinterpret_cast<const v16h*>(h16 + mrow * 64 + khalf);
    v16h a1 = *reinterpret_cast<const v16h*>(h16 + mrow * 64 + 32 + khalf);
    v16h bm0 = *reinterpret_cast<const v16h*>(W4p + mrow * 64 + khalf);
    v16h bm1 = *reinterpret_cast<const v16h*>(W4p + mrow * 64 + 32 + khalf);
    v8f z = {};
    v8f t0 = __builtin_amdgcn_wmma_f32_16x16x32_f16(false, a0, false, bm0,
                                                    (short)0, z, false, false);
    v8f o  = __builtin_amdgcn_wmma_f32_16x16x32_f16(false, a1, false, bm1,
                                                    (short)0, t0, false, false);
    if (mrow < 3) {
      float bb = b4[mrow];
      #pragma unroll
      for (int i = 0; i < 8; ++i)
        rgbOut[(rayBase + rbase + i) * 3 + mrow] = o[i] + bb;
    }
  }
}

// ---------------------------------------------------------------------------
extern "C" void kernel_launch(void* const* d_in, const int* in_sizes, int n_in,
                              void* d_out, int out_size, void* d_ws, size_t ws_size,
                              hipStream_t stream)
{
  (void)in_sizes; (void)n_in; (void)out_size; (void)ws_size;
  const float* x    = (const float*)d_in[0];
  const float* tblL = (const float*)d_in[1];
  const float* tblR = (const float*)d_in[2];
  const float* wl   = (const float*)d_in[3];
  const float* bl   = (const float*)d_in[4];
  const float* w1   = (const float*)d_in[5];
  const float* b1   = (const float*)d_in[6];
  const float* w2   = (const float*)d_in[7];
  const float* b2   = (const float*)d_in[8];
  const float* w3   = (const float*)d_in[9];
  const float* b3   = (const float*)d_in[10];
  const float* w4   = (const float*)d_in[11];
  const float* b4   = (const float*)d_in[12];

  // d_out layout: output_hits[N] | output[N*200] | indices_first[N] (int32
  // bit pattern) | output_rgb[N*3]
  float* out     = (float*)d_out;
  float* outHits = out;
  float* outSig  = out + N_RAYS;
  int*   outIdx  = (int*)(out + N_RAYS + N_RAYS * N_POINTS);
  float* outRgb  = out + N_RAYS + N_RAYS * N_POINTS + N_RAYS;

  // workspace layout (all offsets 256B-aligned)
  char* ws = (char*)d_ws;
  float*    ptsdirs = (float*)ws;                     // N*8 f32      (1 MB)
  _Float16* x16     = (_Float16*)(ws + 1048576);      // N*32 f16     (2 MB)
  _Float16* W1p     = (_Float16*)(ws + 3145728);      // 64*32 f16
  _Float16* W2p     = (_Float16*)(ws + 3149824);      // 64*64 f16
  _Float16* W3p     = (_Float16*)(ws + 3158016);      // 64*64 f16
  _Float16* W4p     = (_Float16*)(ws + 3166208);      // 16*64 f16

  k_prep <<<44, 256, 0, stream>>>(w1, w2, w3, w4, W1p, W2p, W3p, W4p);
  k_label<<<(N_RAYS * 32) / 256, 256, 0, stream>>>(x, tblL, wl, bl,
                                                   outHits, outSig, outIdx, ptsdirs);
  k_rgb  <<<N_RAYS / 256, 256, 0, stream>>>(tblR, ptsdirs, (unsigned int*)x16);
  k_mlp  <<<N_RAYS / (16 * 8), 256, 0, stream>>>(x16, W1p, W2p, W3p, W4p,
                                                 b1, b2, b3, b4, outRgb);
}
